// DCNv2_PyTorch_9285719294376
// MI455X (gfx1250) — matrix operand order repaired and checked by reading, not compiled, not run on hardware
//
#include <hip/hip_runtime.h>
#include <hip/hip_bf16.h>

// ---------------- problem constants (from reference) ----------------
#define Cc    128            // input channels
#define Oo    128            // output channels
#define Hh    64
#define Ww    64
#define Bb    8
#define K2c   9              // 3x3 kernel taps
#define NPIX  4096           // Hh*Ww (= Ho*Wo, stride 1 pad 1)
#define KDIM  (Cc * K2c)     // 1152 : GEMM reduction dim
#define NT    64             // output pixels per tile
#define KC    32             // K-chunk (matches WMMA bf16 K=32)
#define NKC   (KDIM / KC)    // 36 chunks

typedef __bf16 bf16_t;
typedef __attribute__((ext_vector_type(16))) __bf16 v16bf;
typedef __attribute__((ext_vector_type(8)))  float  v8f;

union Frag {
    v16bf v;
    uint4 q[2];
};

__device__ __forceinline__ int clampi(int v, int lo, int hi) {
    return v < lo ? lo : (v > hi ? hi : v);
}
__device__ __forceinline__ float sigmoidf_(float v) {
    return 1.0f / (1.0f + __expf(-v));
}

// ---------------- kernel 1: weight f32 -> bf16 ----------------
__global__ void cvt_weight_bf16(const float* __restrict__ w,
                                bf16_t* __restrict__ wb, int n) {
    int i = blockIdx.x * blockDim.x + threadIdx.x;
    if (i < n) wb[i] = (bf16_t)w[i];
}

// ---------------- kernel 2: offset/mask 3x3 convs ----------------
// One thread per output pixel; 27 accumulators (18 offset ch + 9 mask ch).
// Weights staged in LDS in c-chunks of 32 (27*32*9 floats = 31KB), read as
// wave-uniform broadcasts.
__global__ __launch_bounds__(256) void conv_offset_mask(
    const float* __restrict__ x,
    const float* __restrict__ w_off, const float* __restrict__ b_off,
    const float* __restrict__ w_msk, const float* __restrict__ b_msk,
    float* __restrict__ offs, float* __restrict__ mask)
{
    __shared__ float wl[27 * 32 * 9];   // [oc][c_local][tap]

    const int p   = blockIdx.x * 256 + threadIdx.x;   // pixel id
    const int b   = p >> 12;
    const int rem = p & 4095;
    const int ho  = rem >> 6;
    const int wo  = rem & 63;

    float acc[27];
#pragma unroll
    for (int oc = 0; oc < 18; ++oc) acc[oc] = b_off[oc];
#pragma unroll
    for (int oc = 0; oc < 9;  ++oc) acc[18 + oc] = b_msk[oc];

    for (int cc = 0; cc < Cc; cc += 32) {
        __syncthreads();
        for (int i = threadIdx.x; i < 27 * 32 * 9; i += 256) {
            int oc = i / 288;
            int r  = i - oc * 288;               // c_local*9 + tap
            wl[i] = (oc < 18) ? w_off[oc * KDIM + cc * 9 + r]
                              : w_msk[(oc - 18) * KDIM + cc * 9 + r];
        }
        __syncthreads();

        for (int cl = 0; cl < 32; ++cl) {
            const int c = cc + cl;
            const float* xp = x + (((size_t)(b * Cc + c)) << 12);
            float xv[9];
#pragma unroll
            for (int t = 0; t < 9; ++t) {
                int ky = t / 3, kx = t - 3 * (t / 3);
                int y  = ho - 1 + ky;
                int xx = wo - 1 + kx;
                bool ok = ((unsigned)y < 64u) && ((unsigned)xx < 64u);
                xv[t] = ok ? xp[(y << 6) + xx] : 0.0f;
            }
#pragma unroll
            for (int oc = 0; oc < 27; ++oc) {
                const float* wp = &wl[oc * 288 + cl * 9];
#pragma unroll
                for (int t = 0; t < 9; ++t)
                    acc[oc] = fmaf(wp[t], xv[t], acc[oc]);
            }
        }
    }

#pragma unroll
    for (int oc = 0; oc < 18; ++oc)
        offs[(((size_t)(b * 18 + oc)) << 12) + rem] = acc[oc];
#pragma unroll
    for (int oc = 0; oc < 9; ++oc)
        mask[(((size_t)(b * 9 + oc)) << 12) + rem] = sigmoidf_(acc[18 + oc]);
}

// ---------------- kernel 3: deformable sampling + WMMA GEMM ----------------
// Grid: (64 pixel-tiles, 8 batches). Block: 256 threads = 8 waves.
// Each workgroup produces out[b, 0:128, tile*64 : tile*64+64].
__global__ __launch_bounds__(256) void dcn_wmma(
    const float* __restrict__ x, const bf16_t* __restrict__ wb,
    const float* __restrict__ offs, const float* __restrict__ maskb,
    float* __restrict__ out)
{
    __shared__ __align__(16) bf16_t Alds[Oo * KC];   // 128x32 bf16, row-major (8KB)
    __shared__ __align__(16) bf16_t Blds[NT * KC];   // stored TRANSPOSED: [n][k] (4KB)
    __shared__ __align__(16) int4   sIdx[K2c * NT];  // 4 gather indices per (k2,n)
    __shared__ __align__(16) float4 sWt [K2c * NT];  // 4 bilinear*mask*valid weights

    const int ntile = blockIdx.x;          // 0..63
    const int b     = blockIdx.y;          // 0..7
    const int tid   = threadIdx.x;
    const int lane  = tid & 31;
    const int wave  = tid >> 5;            // 0..7 -> owns M rows [wave*16, wave*16+16)

    // ---- phase 0: bilinear gather descriptors, reused across all 128 c ----
    for (int e = tid; e < K2c * NT; e += 256) {
        int k2 = e >> 6, n = e & 63;
        int ng = (ntile << 6) + n;
        int ho = ng >> 6, wo = ng & 63;
        int ky = k2 / 3, kx = k2 - 3 * ky;

        float dy = offs[(((size_t)(b * 18 + 2 * k2    )) << 12) + ng];
        float dx = offs[(((size_t)(b * 18 + 2 * k2 + 1)) << 12) + ng];
        float m  = maskb[(((size_t)(b * 9 + k2)) << 12) + ng];

        float py = dy + (float)(ky + ho - 1);
        float px = dx + (float)(kx + wo - 1);
        float y0f = floorf(py), x0f = floorf(px);
        float ly = py - y0f, lx = px - x0f;
        int y0 = (int)y0f, x0 = (int)x0f;
        int y1 = y0 + 1,   x1 = x0 + 1;

        float vy0 = ((unsigned)y0 < 64u) ? 1.f : 0.f;
        float vy1 = ((unsigned)y1 < 64u) ? 1.f : 0.f;
        float vx0 = ((unsigned)x0 < 64u) ? 1.f : 0.f;
        float vx1 = ((unsigned)x1 < 64u) ? 1.f : 0.f;
        int y0c = clampi(y0, 0, 63), y1c = clampi(y1, 0, 63);
        int x0c = clampi(x0, 0, 63), x1c = clampi(x1, 0, 63);

        int4 id;
        id.x = (y0c << 6) + x0c;  id.y = (y0c << 6) + x1c;
        id.z = (y1c << 6) + x0c;  id.w = (y1c << 6) + x1c;
        float4 w;
        w.x = (1.f - ly) * (1.f - lx) * vy0 * vx0 * m;
        w.y = (1.f - ly) * lx         * vy0 * vx1 * m;
        w.z = ly * (1.f - lx)         * vy1 * vx0 * m;
        w.w = ly * lx                 * vy1 * vx1 * m;
        sIdx[e] = id;
        sWt[e]  = w;
    }

    v8f acc0 = {}, acc1 = {}, acc2 = {}, acc3 = {};

    const int n_b  = tid & 63;   // B-fill: column
    const int kg_b = tid >> 6;   // B-fill: k-group (0..3) of 8
    const int o_a  = tid >> 1;   // A-fill: weight row
    const int ha_a = tid & 1;    // A-fill: which 16-element half

    for (int kc = 0; kc < NKC; ++kc) {
        __syncthreads();   // previous iteration's fragment reads done (+ phase 0)

        // fill A tile: 16 bf16 (32B) per thread, 16B-aligned global loads
        {
            const bf16_t* src = wb + o_a * KDIM + kc * KC + ha_a * 16;
            uint4 q0 = *(const uint4*)(src);
            uint4 q1 = *(const uint4*)(src + 8);
            *(uint4*)&Alds[o_a * KC + ha_a * 16]     = q0;
            *(uint4*)&Alds[o_a * KC + ha_a * 16 + 8] = q1;
        }
        // fill B tile: 8 deformable samples per thread (4-pt gather each)
        {
            union { bf16_t h[8]; uint4 q; } bv;
#pragma unroll
            for (int j = 0; j < 8; ++j) {
                int k  = kc * KC + kg_b * 8 + j;
                int c  = k / 9;
                int k2 = k - 9 * c;
                int4   id = sIdx[(k2 << 6) + n_b];
                float4 w  = sWt [(k2 << 6) + n_b];
                const float* xp = x + (((size_t)(b * Cc + c)) << 12);
                float v = w.x * xp[id.x] + w.y * xp[id.y]
                        + w.z * xp[id.z] + w.w * xp[id.w];
                bv.h[j] = (bf16_t)v;
            }
            *(uint4*)&Blds[n_b * KC + kg_b * 8] = bv.q;
        }
        __syncthreads();

        // ---- fragments (contiguous k = half*16 + j, same in A and B) ----
        const int half = lane >> 4;
        const int l16  = lane & 15;

        Frag a;   // A 16x32: row = lane%16; elems j -> K = half*16 + j
        {
            int m = (wave << 4) + l16;
            a.q[0] = *(const uint4*)&Alds[m * KC + half * 16];
            a.q[1] = *(const uint4*)&Alds[m * KC + half * 16 + 8];
        }
        // B 32x16: col = lane%16; elems j -> K = half*16 + j (contiguous in Blds[n][k])
        Frag b0, b1, b2, b3;
        b0.q[0] = *(const uint4*)&Blds[( 0 + l16) * KC + half * 16];
        b0.q[1] = *(const uint4*)&Blds[( 0 + l16) * KC + half * 16 + 8];
        b1.q[0] = *(const uint4*)&Blds[(16 + l16) * KC + half * 16];
        b1.q[1] = *(const uint4*)&Blds[(16 + l16) * KC + half * 16 + 8];
        b2.q[0] = *(const uint4*)&Blds[(32 + l16) * KC + half * 16];
        b2.q[1] = *(const uint4*)&Blds[(32 + l16) * KC + half * 16 + 8];
        b3.q[0] = *(const uint4*)&Blds[(48 + l16) * KC + half * 16];
        b3.q[1] = *(const uint4*)&Blds[(48 + l16) * KC + half * 16 + 8];

        acc0 = __builtin_amdgcn_wmma_f32_16x16x32_bf16(false, a.v, false, b0.v,
                                                       (short)0, acc0, false, false);
        acc1 = __builtin_amdgcn_wmma_f32_16x16x32_bf16(false, a.v, false, b1.v,
                                                       (short)0, acc1, false, false);
        acc2 = __builtin_amdgcn_wmma_f32_16x16x32_bf16(false, a.v, false, b2.v,
                                                       (short)0, acc2, false, false);
        acc3 = __builtin_amdgcn_wmma_f32_16x16x32_bf16(false, a.v, false, b3.v,
                                                       (short)0, acc3, false, false);
    }

    // ---- epilogue: C/D layout -> rows 0..7 lanes 0-15, rows 8..15 lanes 16-31
    const int m0 = (wave << 4) + ((lane >> 4) << 3);
    const int nc = (ntile << 6) + (lane & 15);
#pragma unroll
    for (int r = 0; r < 8; ++r) {
        size_t rowbase = ((size_t)(b * Oo + m0 + r)) << 12;
        out[rowbase + nc]      = acc0[r];
        out[rowbase + nc + 16] = acc1[r];
        out[rowbase + nc + 32] = acc2[r];
        out[rowbase + nc + 48] = acc3[r];
    }
}

// ---------------- launch ----------------
extern "C" void kernel_launch(void* const* d_in, const int* in_sizes, int n_in,
                              void* d_out, int out_size, void* d_ws, size_t ws_size,
                              hipStream_t stream) {
    const float* x      = (const float*)d_in[0];
    const float* w_off  = (const float*)d_in[1];
    const float* b_off  = (const float*)d_in[2];
    const float* w_msk  = (const float*)d_in[3];
    const float* b_msk  = (const float*)d_in[4];
    const float* weight = (const float*)d_in[5];
    float* out = (float*)d_out;

    char* ws = (char*)d_ws;
    float*  offs = (float*)ws;                                        // 8*18*4096 f32
    float*  mask = (float*)(ws + (size_t)Bb * 18 * NPIX * 4);         // 8*9*4096 f32
    bf16_t* wbf  = (bf16_t*)(ws + (size_t)Bb * 27 * NPIX * 4);        // 128*1152 bf16

    cvt_weight_bf16<<<(Oo * KDIM + 255) / 256, 256, 0, stream>>>(weight, wbf, Oo * KDIM);
    conv_offset_mask<<<(Bb * NPIX) / 256, 256, 0, stream>>>(x, w_off, b_off,
                                                            w_msk, b_msk, offs, mask);
    dim3 g2(NPIX / NT, Bb);
    dcn_wmma<<<g2, 256, 0, stream>>>(x, wbf, offs, mask, out);
}
